// GCCGraphInfer_48215302864932
// MI455X (gfx1250) — matrix operand order repaired and checked by reading, not compiled, not run on hardware
//
#include <hip/hip_runtime.h>
#include <hip/hip_bf16.h>
#include <math.h>

#define N_NODES 50000
#define N_EDGES 500000
#define N_GRAPH 64
#define DIN 128
#define EDIM 32
#define H1 256
#define H2 128
#define HD 64
#define OUTD 8

typedef __bf16 v16bf __attribute__((ext_vector_type(16)));
typedef __bf16 v8bf  __attribute__((ext_vector_type(8)));
typedef float  v8f   __attribute__((ext_vector_type(8)));

// ---- monotonic float<->uint key for atomic segment-max -------------------
__device__ __forceinline__ unsigned f32_key(float f) {
  unsigned u = __float_as_uint(f);
  return ((int)u >= 0) ? (u | 0x80000000u) : ~u;
}
__device__ __forceinline__ float key_f32(unsigned k) {
  unsigned u = (k & 0x80000000u) ? (k & 0x7fffffffu) : ~k;
  return __uint_as_float(u);
}

// ---- precision staging ----------------------------------------------------
__global__ void cvt_f32_bf16(const float* __restrict__ X, __bf16* __restrict__ Y, int total) {
  int idx = blockIdx.x * blockDim.x + threadIdx.x;
  if (idx >= total) return;
  Y[idx] = (__bf16)X[idx];
}

// W[K x H] f32 row-major -> Wt[H x K] bf16 (column-major weights)
__global__ void cvt_transpose_bf16(const float* __restrict__ W, __bf16* __restrict__ Wt,
                                   int K, int H) {
  int idx = blockIdx.x * blockDim.x + threadIdx.x;
  if (idx >= K * H) return;
  int k = idx / H, n = idx % H;
  Wt[(size_t)n * K + k] = (__bf16)W[idx];
}

// ---- WMMA bf16 GEMM: C[rows x H] = Xb[rows x K] @ Wt^T + bias ------------
// One wave owns a 16(M) x 64(N) tile: one A fragment feeds 4 WMMAs.
// A frag (ISA 7.12.2, 16-bit A 16x32): lane m=l&15; K runs [k0+8*half, +8) and
//   [k0+16+8*half, +8)  -> two contiguous b128 loads from bf16 row.
// B frag: lane col n=l&15; K runs [k0+8*half, +8) and
//   [k0+16+8*half, +8) (same as A) -> two b128 loads from Wt column.
template <int K, int H>
__global__ void gemm_bias_wmma(const __bf16* __restrict__ Xb,
                               const __bf16* __restrict__ Wt,
                               const float* __restrict__ bias,
                               float* __restrict__ C, int rows) {
  const int lane   = threadIdx.x & 31;
  const int wib    = threadIdx.x >> 5;          // 8 waves / block
  const int tilesN = H / 64;
  const int tile   = blockIdx.x * 8 + wib;
  const int total  = (rows >> 4) * tilesN;
  if (tile >= total) return;                    // wave-uniform: EXEC all-1 inside
  const int tm   = tile / tilesN;
  const int tn   = tile % tilesN;
  const int half = lane >> 4;
  const int mn   = lane & 15;

  const __bf16* xrow = Xb + (size_t)(tm * 16 + mn) * K;

  v8f acc[4];
#pragma unroll
  for (int s = 0; s < 4; ++s) acc[s] = (v8f){};

#pragma unroll
  for (int k0 = 0; k0 < K; k0 += 32) {
    const int kbA = k0 + half * 8;
    v8bf alo = *(const v8bf*)(xrow + kbA);
    v8bf ahi = *(const v8bf*)(xrow + kbA + 16);
    v16bf a = __builtin_shufflevector(alo, ahi, 0, 1, 2, 3, 4, 5, 6, 7,
                                      8, 9, 10, 11, 12, 13, 14, 15);
    const int kbB = k0 + half * 8;
#pragma unroll
    for (int s = 0; s < 4; ++s) {
      const __bf16* wcol = Wt + (size_t)(tn * 64 + s * 16 + mn) * K;
      v8bf blo = *(const v8bf*)(wcol + kbB);
      v8bf bhi = *(const v8bf*)(wcol + kbB + 16);
      v16bf b = __builtin_shufflevector(blo, bhi, 0, 1, 2, 3, 4, 5, 6, 7,
                                        8, 9, 10, 11, 12, 13, 14, 15);
      acc[s] = __builtin_amdgcn_wmma_f32_16x16x32_bf16(false, a, false, b,
                                                       (short)0, acc[s], false, false);
    }
  }

#pragma unroll
  for (int s = 0; s < 4; ++s) {
    const int col = tn * 64 + s * 16 + mn;
    const float bv = bias[col];
#pragma unroll
    for (int r = 0; r < 8; ++r) {
      const int m = tm * 16 + r + half * 8;     // C/D: VGPR r -> M=r (+8 hi lanes)
      C[(size_t)m * H + col] = acc[s][r] + bv;
    }
  }
}

// ---- self-loop edge_attr: segment mean over dst --------------------------
__global__ void loop_accum(const float* __restrict__ eattr,
                           const int* __restrict__ dst,
                           float* lattr, float* deg) {
  int idx = blockIdx.x * blockDim.x + threadIdx.x;
  if (idx >= N_EDGES * EDIM) return;
  int e = idx >> 5, k = idx & 31;
  int d = dst[e];
  atomicAdd(lattr + (size_t)d * EDIM + k, eattr[idx]);
  if (k == 0) atomicAdd(deg + d, 1.0f);
}

__global__ void loop_final(float* lattr, const float* __restrict__ deg) {
  int idx = blockIdx.x * blockDim.x + threadIdx.x;
  if (idx >= N_NODES * EDIM) return;
  lattr[idx] /= fmaxf(deg[idx >> 5], 1.0f);
}

// ---- GATv2 edge score: e = att . leakyrelu(xl[s] + xr[d] + eattr@We) -----
// One wave per (edge | self-loop). EDIM==32 == wave32: eattr broadcast by shuffle.
__global__ void edge_score(const float* __restrict__ xl, const float* __restrict__ xr,
                           const float* __restrict__ eattr, const float* __restrict__ lattr,
                           const int* __restrict__ src, const int* __restrict__ dst,
                           const float* __restrict__ We, const float* __restrict__ att,
                           float* score, unsigned* mkey, int H) {
  int idx = blockIdx.x * blockDim.x + threadIdx.x;
  int e = idx >> 5, lane = idx & 31;
  if (e >= N_EDGES + N_NODES) return;
  int s, d; const float* ea;
  if (e < N_EDGES) { s = src[e]; d = dst[e]; ea = eattr + (size_t)e * EDIM; }
  else             { s = d = e - N_EDGES;    ea = lattr + (size_t)(e - N_EDGES) * EDIM; }
  float ev = ea[lane];
  float acc = 0.f;
  const int cpl = H >> 5;                    // channels per lane
  for (int j = 0; j < cpl; ++j) {
    int c = j * 32 + lane;
    float t = xl[(size_t)s * H + c] + xr[(size_t)d * H + c];
    float w = 0.f;
#pragma unroll
    for (int k = 0; k < EDIM; ++k)
      w += __shfl(ev, k, 32) * We[k * H + c];
    t += w;
    t = (t > 0.f) ? t : 0.2f * t;            // leaky_relu(., 0.2)
    acc += att[c] * t;
  }
#pragma unroll
  for (int off = 16; off > 0; off >>= 1) acc += __shfl_down(acc, off, 32);
  if (lane == 0) {
    score[e] = acc;
    atomicMax(mkey + d, f32_key(acc));       // segment-max via ordered uint key
  }
}

// ---- softmax weights + denominator ---------------------------------------
__global__ void edge_exp(float* score, const unsigned* __restrict__ mkey,
                         const int* __restrict__ dst, float* denom) {
  int e = blockIdx.x * blockDim.x + threadIdx.x;
  if (e >= N_EDGES + N_NODES) return;
  int d = (e < N_EDGES) ? dst[e] : (e - N_EDGES);
  float w = expf(score[e] - key_f32(mkey[d]));
  score[e] = w;
  atomicAdd(denom + d, w);
}

// ---- weighted scatter-aggregate: out[d] += alpha * xl[s] -----------------
__global__ void edge_aggregate(const float* __restrict__ xl,
                               const float* __restrict__ score,
                               const float* __restrict__ denom,
                               const int* __restrict__ src, const int* __restrict__ dst,
                               float* out, int H) {
  int idx = blockIdx.x * blockDim.x + threadIdx.x;
  int e = idx >> 5, lane = idx & 31;
  if (e >= N_EDGES + N_NODES) return;
  int s, d;
  if (e < N_EDGES) { s = src[e]; d = dst[e]; } else { s = d = e - N_EDGES; }
  float alpha = score[e] / denom[d];
  const int cpl = H >> 5;
  for (int j = 0; j < cpl; ++j) {
    int c = j * 32 + lane;
    atomicAdd(out + (size_t)d * H + c, alpha * xl[(size_t)s * H + c]);
  }
}

__global__ void bias_relu(float* x, const float* __restrict__ b, int total, int H) {
  int idx = blockIdx.x * blockDim.x + threadIdx.x;
  if (idx >= total) return;
  float v = x[idx] + b[idx % H];
  x[idx] = v > 0.f ? v : 0.f;
}

// ---- global mean pool -----------------------------------------------------
__global__ void graph_count(const int* __restrict__ batch, float* cnt) {
  int n = blockIdx.x * blockDim.x + threadIdx.x;
  if (n >= N_NODES) return;
  atomicAdd(cnt + batch[n], 1.0f);
}

__global__ void graph_pool(const float* __restrict__ x, const int* __restrict__ batch,
                           float* pooled) {
  int idx = blockIdx.x * blockDim.x + threadIdx.x;
  if (idx >= N_NODES * H2) return;
  int n = idx / H2, c = idx % H2;
  atomicAdd(pooled + (size_t)batch[n] * H2 + c, x[idx]);
}

// ---- head: dense + BN(eval) + leaky(0.1), then dense -> out --------------
__global__ void dense_bn(const float* __restrict__ pooled, const float* __restrict__ cnt,
                         const float* __restrict__ Wd1, const float* __restrict__ bd1,
                         const float* __restrict__ gamma, const float* __restrict__ beta,
                         const float* __restrict__ mean, const float* __restrict__ var,
                         float* hidden) {
  int idx = blockIdx.x * blockDim.x + threadIdx.x;
  if (idx >= N_GRAPH * HD) return;
  int g = idx / HD, h = idx % HD;
  float inv = 1.0f / fmaxf(cnt[g], 1.0f);
  float s = bd1[h];
  for (int k = 0; k < H2; ++k) s += pooled[g * H2 + k] * inv * Wd1[k * HD + h];
  float v = (s - mean[h]) * rsqrtf(var[h] + 1e-5f) * gamma[h] + beta[h];
  hidden[idx] = v > 0.f ? v : 0.1f * v;
}

__global__ void dense_out(const float* __restrict__ hidden,
                          const float* __restrict__ Wd2, const float* __restrict__ bd2,
                          float* out) {
  int idx = blockIdx.x * blockDim.x + threadIdx.x;
  if (idx >= N_GRAPH * OUTD) return;
  int g = idx / OUTD, o = idx % OUTD;
  float s = bd2[o];
  for (int k = 0; k < HD; ++k) s += hidden[g * HD + k] * Wd2[k * OUTD + o];
  out[idx] = s;
}

extern "C" void kernel_launch(void* const* d_in, const int* in_sizes, int n_in,
                              void* d_out, int out_size, void* d_ws, size_t ws_size,
                              hipStream_t stream) {
  const float* node_attr = (const float*)d_in[0];
  const float* edge_attr = (const float*)d_in[1];
  const int*   edge_src  = (const int*)d_in[2];
  const int*   edge_dst  = (const int*)d_in[3];
  const int*   batch     = (const int*)d_in[4];
  const float* Wl1 = (const float*)d_in[5];  const float* bl1 = (const float*)d_in[6];
  const float* Wr1 = (const float*)d_in[7];  const float* br1 = (const float*)d_in[8];
  const float* We1 = (const float*)d_in[9];  const float* att1 = (const float*)d_in[10];
  const float* b1  = (const float*)d_in[11];
  const float* Wl2 = (const float*)d_in[12]; const float* bl2 = (const float*)d_in[13];
  const float* Wr2 = (const float*)d_in[14]; const float* br2 = (const float*)d_in[15];
  const float* We2 = (const float*)d_in[16]; const float* att2 = (const float*)d_in[17];
  const float* b2  = (const float*)d_in[18];
  const float* Wd1 = (const float*)d_in[19]; const float* bd1 = (const float*)d_in[20];
  const float* bn_g = (const float*)d_in[21]; const float* bn_b = (const float*)d_in[22];
  const float* bn_m = (const float*)d_in[23]; const float* bn_v = (const float*)d_in[24];
  const float* Wd2 = (const float*)d_in[25]; const float* bd2 = (const float*)d_in[26];
  float* out = (float*)d_out;

  // ---- workspace layout (~180 MB) ----
  float* ws = (float*)d_ws;
  size_t off = 0;
  float* xl    = ws + off; off += (size_t)N_NODES * H1;   // projections (H1 or H2 wide)
  float* xr    = ws + off; off += (size_t)N_NODES * H1;
  float* xbuf  = ws + off; off += (size_t)N_NODES * H1;   // agg output / layer input
  float* lattr = ws + off; off += (size_t)N_NODES * EDIM; // self-loop edge attr
  float* deg   = ws + off; off += N_NODES;
  unsigned* mkey = (unsigned*)(ws + off); off += N_NODES;
  float* denom = ws + off; off += N_NODES;
  float* score = ws + off; off += (N_EDGES + N_NODES);
  float* cnt   = ws + off; off += N_GRAPH;
  float* pooled = ws + off; off += (size_t)N_GRAPH * H2;
  float* hidden = ws + off; off += (size_t)N_GRAPH * HD;
  __bf16* Xb   = (__bf16*)(ws + off); off += (size_t)N_NODES * H1 / 2; // bf16 input stage
  __bf16* WtA  = (__bf16*)(ws + off); off += (H1 * H2) / 2;            // bf16 weight (col-major)
  __bf16* WtB  = (__bf16*)(ws + off); off += (H1 * H2) / 2;

  const int EW = N_EDGES + N_NODES;                 // edges incl. self-loops
  const int TB = 256;

  // ---- self-loop edge attributes (segment mean) ----
  hipMemsetAsync(lattr, 0, (size_t)N_NODES * EDIM * 4, stream);
  hipMemsetAsync(deg,   0, (size_t)N_NODES * 4, stream);
  loop_accum<<<(N_EDGES * EDIM + TB - 1) / TB, TB, 0, stream>>>(edge_attr, edge_dst, lattr, deg);
  loop_final<<<(N_NODES * EDIM + TB - 1) / TB, TB, 0, stream>>>(lattr, deg);

  // ================= Layer 1 (K=DIN=128, H=256) =================
  {
    cvt_f32_bf16<<<(N_NODES * DIN + TB - 1) / TB, TB, 0, stream>>>(node_attr, Xb, N_NODES * DIN);
    cvt_transpose_bf16<<<(DIN * H1 + TB - 1) / TB, TB, 0, stream>>>(Wl1, WtA, DIN, H1);
    cvt_transpose_bf16<<<(DIN * H1 + TB - 1) / TB, TB, 0, stream>>>(Wr1, WtB, DIN, H1);
    int tiles = (N_NODES / 16) * (H1 / 64);
    gemm_bias_wmma<DIN, H1><<<(tiles + 7) / 8, TB, 0, stream>>>(Xb, WtA, bl1, xl, N_NODES);
    gemm_bias_wmma<DIN, H1><<<(tiles + 7) / 8, TB, 0, stream>>>(Xb, WtB, br1, xr, N_NODES);
    hipMemsetAsync(mkey,  0, (size_t)N_NODES * 4, stream);
    hipMemsetAsync(denom, 0, (size_t)N_NODES * 4, stream);
    hipMemsetAsync(xbuf,  0, (size_t)N_NODES * H1 * 4, stream);
    edge_score<<<((size_t)EW * 32 + TB - 1) / TB, TB, 0, stream>>>(
        xl, xr, edge_attr, lattr, edge_src, edge_dst, We1, att1, score, mkey, H1);
    edge_exp<<<(EW + TB - 1) / TB, TB, 0, stream>>>(score, mkey, edge_dst, denom);
    edge_aggregate<<<((size_t)EW * 32 + TB - 1) / TB, TB, 0, stream>>>(
        xl, score, denom, edge_src, edge_dst, xbuf, H1);
    bias_relu<<<(N_NODES * H1 + TB - 1) / TB, TB, 0, stream>>>(xbuf, b1, N_NODES * H1, H1);
  }

  // ================= Layer 2 (K=H1=256, H=128) =================
  {
    cvt_f32_bf16<<<(N_NODES * H1 + TB - 1) / TB, TB, 0, stream>>>(xbuf, Xb, N_NODES * H1);
    cvt_transpose_bf16<<<(H1 * H2 + TB - 1) / TB, TB, 0, stream>>>(Wl2, WtA, H1, H2);
    cvt_transpose_bf16<<<(H1 * H2 + TB - 1) / TB, TB, 0, stream>>>(Wr2, WtB, H1, H2);
    int tiles = (N_NODES / 16) * (H2 / 64);
    gemm_bias_wmma<H1, H2><<<(tiles + 7) / 8, TB, 0, stream>>>(Xb, WtA, bl2, xl, N_NODES);
    gemm_bias_wmma<H1, H2><<<(tiles + 7) / 8, TB, 0, stream>>>(Xb, WtB, br2, xr, N_NODES);
    hipMemsetAsync(mkey,  0, (size_t)N_NODES * 4, stream);
    hipMemsetAsync(denom, 0, (size_t)N_NODES * 4, stream);
    hipMemsetAsync(xbuf,  0, (size_t)N_NODES * H2 * 4, stream);   // x1 consumed; reuse as x2
    edge_score<<<((size_t)EW * 32 + TB - 1) / TB, TB, 0, stream>>>(
        xl, xr, edge_attr, lattr, edge_src, edge_dst, We2, att2, score, mkey, H2);
    edge_exp<<<(EW + TB - 1) / TB, TB, 0, stream>>>(score, mkey, edge_dst, denom);
    edge_aggregate<<<((size_t)EW * 32 + TB - 1) / TB, TB, 0, stream>>>(
        xl, score, denom, edge_src, edge_dst, xbuf, H2);
    bias_relu<<<(N_NODES * H2 + TB - 1) / TB, TB, 0, stream>>>(xbuf, b2, N_NODES * H2, H2);
  }

  // ================= Pool + head =================
  hipMemsetAsync(cnt,    0, (size_t)N_GRAPH * 4, stream);
  hipMemsetAsync(pooled, 0, (size_t)N_GRAPH * H2 * 4, stream);
  graph_count<<<(N_NODES + TB - 1) / TB, TB, 0, stream>>>(batch, cnt);
  graph_pool<<<(N_NODES * H2 + TB - 1) / TB, TB, 0, stream>>>(xbuf, batch, pooled);
  dense_bn<<<(N_GRAPH * HD + TB - 1) / TB, TB, 0, stream>>>(
      pooled, cnt, Wd1, bd1, bn_g, bn_b, bn_m, bn_v, hidden);
  dense_out<<<(N_GRAPH * OUTD + TB - 1) / TB, TB, 0, stream>>>(hidden, Wd2, bd2, out);
}
